// GCNPolicySpeedRoute_18811956756658
// MI455X (gfx1250) — matrix operand order repaired and checked by reading, not compiled, not run on hardware
//
#include <hip/hip_runtime.h>
#include <hip/hip_bf16.h>

typedef __attribute__((ext_vector_type(16))) __bf16 v16bf;
typedef __attribute__((ext_vector_type(8)))  float  v8f;

#define HCOLS 64
#define EPSBN 1e-5f

static inline int cdiv(long long a, long long b) { return (int)((a + b - 1) / b); }

// ---------------------------------------------------------------------------
// Utility fill
// ---------------------------------------------------------------------------
__global__ void k_fill(float* __restrict__ p, float v, long long n) {
  long long i = (long long)blockIdx.x * blockDim.x + threadIdx.x;
  if (i < n) p[i] = v;
}

// ---------------------------------------------------------------------------
// Degree accumulation (deg pre-initialized to 1.0 = self loop weight)
// ---------------------------------------------------------------------------
__global__ void k_deg_edges(const int* __restrict__ col, const float* __restrict__ ew,
                            float* __restrict__ deg, int E) {
  int e = blockIdx.x * blockDim.x + threadIdx.x;
  if (e >= E) return;
  atomicAdd(&deg[col[e]], ew[e]);
}

__global__ void k_rsqrt_inplace(float* __restrict__ d, int n) {
  int i = blockIdx.x * blockDim.x + threadIdx.x;
  if (i >= n) return;
  float v = d[i];
  d[i] = (v > 0.f) ? rsqrtf(v) : 0.f;
}

__global__ void k_edge_norm(const int* __restrict__ row, const int* __restrict__ col,
                            const float* __restrict__ ew, const float* __restrict__ dis,
                            float* __restrict__ norm, int E) {
  int e = blockIdx.x * blockDim.x + threadIdx.x;
  if (e >= E) return;
  norm[e] = dis[row[e]] * ew[e] * dis[col[e]];
}

// ---------------------------------------------------------------------------
// WMMA GEMM: C[M x 64] = A[M x K] @ W[K x 64]   (K in {64,128}, M mult of 16)
//
// W is staged once per block into LDS, pre-swizzled into the bf16 B-fragment
// layout (elem i of lane L, tile t, kstep s  ->  K=32s+8*(L>>4)+(i&7)+16*(i>>3), n=16t+(L&15)).
// Each wave grid-strides over 16-row tiles; A fragments via float4 loads,
// B fragments via 32-byte ds_loads, accumulation in v_wmma_f32_16x16x32_bf16.
// (K kept runtime: prevents LICM from hoisting all B fragments into registers,
//  which was shown to spill to scratch for K=128.)
// ---------------------------------------------------------------------------
__global__ void k_gemm_wmma(const float* __restrict__ A, const float* __restrict__ W,
                            float* __restrict__ C, int M, int K) {
  __shared__ __attribute__((aligned(32))) __bf16 wlds[4 * 4 * 32 * 16];  // 16 KB max (K=128)

  const int nk = K >> 5;                     // K-steps of 32
  const int nent = nk << 11;                 // nk*4*32*16 fragment elements
  for (int idx = threadIdx.x; idx < nent; idx += blockDim.x) {
    int i = idx & 15;
    int L = (idx >> 4) & 31;
    int t = (idx >> 9) & 3;
    int s = idx >> 11;
    int kk = 32 * s + 8 * (L >> 4) + (i & 7) + 16 * (i >> 3);
    int n  = 16 * t + (L & 15);
    wlds[idx] = (__bf16)W[(size_t)kk * HCOLS + n];
  }
  __syncthreads();

  const int lane = threadIdx.x & 31;
  const int wave = threadIdx.x >> 5;
  const int wpb  = blockDim.x >> 5;
  const int half = lane >> 4;
  const int l16  = lane & 15;
  const int nt   = M >> 4;                   // number of 16-row tiles
  const int tstride = gridDim.x * wpb;

  for (int tile = blockIdx.x * wpb + wave; tile < nt; tile += tstride) {
    const int row0 = tile * 16;

    // speculative prefetch of the next tile's A rows for this lane
    __builtin_prefetch(A + (size_t)(row0 + tstride * 16 + l16) * K, 0, 1);

    v8f acc[4];
#pragma unroll
    for (int t = 0; t < 4; ++t) acc[t] = (v8f){0.f,0.f,0.f,0.f,0.f,0.f,0.f,0.f};

    for (int s = 0; s < nk; ++s) {
      const int k0 = s * 32;
      // A fragment (16x32, row M=l16): elems 0..7 -> K=k0+8*half+0..7,
      //                                elems 8..15 -> K=k0+8*half+16..23
      const float4* ap = (const float4*)(A + (size_t)(row0 + l16) * K + k0 + 8 * half);
      float4 u0 = ap[0], u1 = ap[1], u2 = ap[4], u3 = ap[5];
      v16bf a;
      a[0]  = (__bf16)u0.x; a[1]  = (__bf16)u0.y; a[2]  = (__bf16)u0.z; a[3]  = (__bf16)u0.w;
      a[4]  = (__bf16)u1.x; a[5]  = (__bf16)u1.y; a[6]  = (__bf16)u1.z; a[7]  = (__bf16)u1.w;
      a[8]  = (__bf16)u2.x; a[9]  = (__bf16)u2.y; a[10] = (__bf16)u2.z; a[11] = (__bf16)u2.w;
      a[12] = (__bf16)u3.x; a[13] = (__bf16)u3.y; a[14] = (__bf16)u3.z; a[15] = (__bf16)u3.w;

#pragma unroll
      for (int t = 0; t < 4; ++t) {
        const v16bf b = *(const v16bf*)&wlds[(((s << 2) + t) << 9) + (lane << 4)];
        acc[t] = __builtin_amdgcn_wmma_f32_16x16x32_bf16(
            false, a, false, b, (short)0, acc[t], false, false);
      }
    }

    // C/D layout: VGPR v holds row (v + 8*half), column l16 of each 16-col tile
#pragma unroll
    for (int t = 0; t < 4; ++t) {
      const int n = t * 16 + l16;
#pragma unroll
      for (int v = 0; v < 8; ++v)
        C[(size_t)(row0 + v + 8 * half) * HCOLS + n] = acc[t][v];
    }
  }
}

// ---------------------------------------------------------------------------
// out[i][f] = b[f] + t[i][f] * dis[i]^2   (self-loop term + bias, zero-base for atomics)
// ---------------------------------------------------------------------------
__global__ void k_init_out(const float* __restrict__ t, const float* __restrict__ dis,
                           const float* __restrict__ bias, float* __restrict__ out,
                           long long total) {
  long long gid = (long long)blockIdx.x * blockDim.x + threadIdx.x;
  if (gid >= total) return;
  int i = (int)(gid >> 6);
  int f = (int)(gid & 63);
  float d = dis[i];
  out[gid] = bias[f] + t[gid] * d * d;
}

// ---------------------------------------------------------------------------
// Edge scatter: out[col[e]][f] += t[row[e]][f] * norm[e]
// ---------------------------------------------------------------------------
__global__ void k_scatter(const float* __restrict__ t, const int* __restrict__ row,
                          const int* __restrict__ col, const float* __restrict__ norm,
                          float* __restrict__ out, long long total) {
  long long gid = (long long)blockIdx.x * blockDim.x + threadIdx.x;
  if (gid >= total) return;
  int e = (int)(gid >> 6);
  int f = (int)(gid & 63);
  float v = t[(size_t)row[e] * HCOLS + f] * norm[e];
  atomicAdd(&out[(size_t)col[e] * HCOLS + f], v);
}

// ---------------------------------------------------------------------------
// Per-column sum / sumsq over rows (64 columns), partials + atomics
// ---------------------------------------------------------------------------
__global__ void k_colstats(const float* __restrict__ x, int Nrows,
                           float* __restrict__ sums, float* __restrict__ sqs) {
  __shared__ float s[256], q[256];
  const int f = threadIdx.x & 63;
  const int rl = threadIdx.x >> 6;             // 0..3
  float sum = 0.f, sq = 0.f;
  for (int r = blockIdx.x * 4 + rl; r < Nrows; r += gridDim.x * 4) {
    float v = x[(size_t)r * HCOLS + f];
    sum += v; sq += v * v;
  }
  s[threadIdx.x] = sum; q[threadIdx.x] = sq;
  __syncthreads();
  if (threadIdx.x < 64) {
    float ts = s[threadIdx.x] + s[threadIdx.x + 64] + s[threadIdx.x + 128] + s[threadIdx.x + 192];
    float tq = q[threadIdx.x] + q[threadIdx.x + 64] + q[threadIdx.x + 128] + q[threadIdx.x + 192];
    atomicAdd(&sums[f], ts);
    atomicAdd(&sqs[f], tq);
  }
}

__global__ void k_bnrelu(const float* __restrict__ x, const float* __restrict__ sums,
                         const float* __restrict__ sqs, const float* __restrict__ g,
                         const float* __restrict__ be, float* __restrict__ h,
                         int Nrows, long long total) {
  long long gid = (long long)blockIdx.x * blockDim.x + threadIdx.x;
  if (gid >= total) return;
  int f = (int)(gid & 63);
  float m = sums[f] / (float)Nrows;
  float var = sqs[f] / (float)Nrows - m * m;
  float v = (x[gid] - m) * rsqrtf(var + EPSBN) * g[f] + be[f];
  h[gid] = fmaxf(v, 0.f);
}

// ---------------------------------------------------------------------------
// segment_max via uint atomicMax (valid: post-ReLU values >= 0, gx zeroed)
// ---------------------------------------------------------------------------
__global__ void k_segmax(const float* __restrict__ h, const int* __restrict__ bidx,
                         unsigned int* __restrict__ gx, long long total) {
  long long gid = (long long)blockIdx.x * blockDim.x + threadIdx.x;
  if (gid >= total) return;
  int i = (int)(gid >> 6);
  unsigned int bits = __float_as_uint(h[gid]);
  atomicMax(&gx[(size_t)bidx[i] * HCOLS + (gid & 63)], bits);
}

// ---------------------------------------------------------------------------
// Speed/route encoders + output MLP head. One block, 64 threads (thread = batch row).
// ---------------------------------------------------------------------------
__global__ void k_head(const float* __restrict__ gx, const float* __restrict__ speed,
                       const float* __restrict__ route,
                       const float* sw, const float* sb, const float* sg, const float* sbe,
                       const float* cw, const float* cb, const float* rg, const float* rbe,
                       const float* rw, const float* rb,
                       const float* ow1, const float* ob1, const float* og, const float* obe,
                       const float* ow2, const float* ob2,
                       float* __restrict__ out) {
  __shared__ float sv[64][4];
  __shared__ float rc[64][10];
  __shared__ float venc[64][4];
  __shared__ float renc[64][4];
  __shared__ float o1[64][16];
  __shared__ float stat[2][16];
  __shared__ float rmv[2];

  const int b = threadIdx.x;   // 0..63

  // speed encoder pre-BN: speed[b] @ sw + sb
  float sp = speed[b];
#pragma unroll
  for (int j = 0; j < 4; ++j) sv[b][j] = sp * sw[j] + sb[j];

  // route conv1d(2->1, k=3, SAME) pre-BN
#pragma unroll
  for (int t = 0; t < 10; ++t) {
    float acc = cb[0];
#pragma unroll
    for (int h = 0; h < 3; ++h) {
      int tt = t + h - 1;
      if (tt >= 0 && tt < 10) {
        acc += route[b * 20 + tt * 2 + 0] * cw[0 * 3 + h];
        acc += route[b * 20 + tt * 2 + 1] * cw[1 * 3 + h];
      }
    }
    rc[b][t] = acc;
  }
  __syncthreads();

  if (b < 4) {                       // speed BN stats (64 rows, 4 cols)
    float s = 0.f, q = 0.f;
    for (int r = 0; r < 64; ++r) { float v = sv[r][b]; s += v; q += v * v; }
    float m = s / 64.f;
    stat[0][b] = m; stat[1][b] = q / 64.f - m * m;
  }
  if (b == 63) {                     // route BN stats (all 640 values)
    float s = 0.f, q = 0.f;
    for (int r = 0; r < 64; ++r)
      for (int t = 0; t < 10; ++t) { float v = rc[r][t]; s += v; q += v * v; }
    float m = s / 640.f;
    rmv[0] = m; rmv[1] = q / 640.f - m * m;
  }
  __syncthreads();

#pragma unroll
  for (int j = 0; j < 4; ++j) {
    float v = (sv[b][j] - stat[0][j]) * rsqrtf(stat[1][j] + EPSBN) * sg[j] + sbe[j];
    venc[b][j] = fmaxf(v, 0.f);
  }
  {
    float rinv = rsqrtf(rmv[1] + EPSBN);
    float rr[10];
#pragma unroll
    for (int t = 0; t < 10; ++t)
      rr[t] = fmaxf((rc[b][t] - rmv[0]) * rinv * rg[0] + rbe[0], 0.f);
#pragma unroll
    for (int j = 0; j < 4; ++j) {
      float a = rb[j];
#pragma unroll
      for (int t = 0; t < 10; ++t) a += rr[t] * rw[t * 4 + j];
      renc[b][j] = a;
    }
  }
  __syncthreads();

  // head linear1: cat = [gx(64) | venc(4) | renc(4)] @ ow1 + ob1
  for (int k = 0; k < 16; ++k) {
    float a = ob1[k];
    for (int i = 0; i < 64; ++i) a += gx[b * 64 + i] * ow1[i * 16 + k];
#pragma unroll
    for (int i = 0; i < 4; ++i)  a += venc[b][i] * ow1[(64 + i) * 16 + k];
#pragma unroll
    for (int i = 0; i < 4; ++i)  a += renc[b][i] * ow1[(68 + i) * 16 + k];
    o1[b][k] = a;
  }
  __syncthreads();
  if (b < 16) {
    float s = 0.f, q = 0.f;
    for (int r = 0; r < 64; ++r) { float v = o1[r][b]; s += v; q += v * v; }
    float m = s / 64.f;
    stat[0][b] = m; stat[1][b] = q / 64.f - m * m;
  }
  __syncthreads();
  float h16[16];
#pragma unroll
  for (int k = 0; k < 16; ++k)
    h16[k] = fmaxf((o1[b][k] - stat[0][k]) * rsqrtf(stat[1][k] + EPSBN) * og[k] + obe[k], 0.f);
#pragma unroll
  for (int a = 0; a < 5; ++a) {
    float o = ob2[a];
#pragma unroll
    for (int k = 0; k < 16; ++k) o += h16[k] * ow2[k * 5 + a];
    out[b * 5 + a] = o;
  }
}

// ---------------------------------------------------------------------------
extern "C" void kernel_launch(void* const* d_in, const int* in_sizes, int n_in,
                              void* d_out, int out_size, void* d_ws, size_t ws_size,
                              hipStream_t stream) {
  const float* x     = (const float*)d_in[0];
  const int*   ei    = (const int*)d_in[1];
  const float* ew    = (const float*)d_in[2];
  const int*   bidx  = (const int*)d_in[3];
  const float* speed = (const float*)d_in[4];
  const float* route = (const float*)d_in[5];
  const float* W1  = (const float*)d_in[6];
  const float* b1  = (const float*)d_in[7];
  const float* g1  = (const float*)d_in[8];
  const float* be1 = (const float*)d_in[9];
  const float* W2  = (const float*)d_in[10];
  const float* b2  = (const float*)d_in[11];
  const float* g2  = (const float*)d_in[12];
  const float* be2 = (const float*)d_in[13];
  const float* sw  = (const float*)d_in[14];
  const float* sb  = (const float*)d_in[15];
  const float* sg  = (const float*)d_in[16];
  const float* sbe = (const float*)d_in[17];
  const float* cw  = (const float*)d_in[18];
  const float* cb  = (const float*)d_in[19];
  const float* rg  = (const float*)d_in[20];
  const float* rbe = (const float*)d_in[21];
  const float* rw  = (const float*)d_in[22];
  const float* rb  = (const float*)d_in[23];
  const float* ow1 = (const float*)d_in[24];
  const float* ob1 = (const float*)d_in[25];
  const float* og  = (const float*)d_in[26];
  const float* obe = (const float*)d_in[27];
  const float* ow2 = (const float*)d_in[28];
  const float* ob2 = (const float*)d_in[29];

  const int E = in_sizes[2];
  const int N = in_sizes[3];
  const int F = in_sizes[0] / N;     // 128
  const int B = in_sizes[4];         // 64
  const int* row = ei;
  const int* col = ei + E;

  const long long N64 = (long long)N * HCOLS;
  float* ws    = (float*)d_ws;
  float* bufA  = ws;                 // GEMM output (pre-scatter)   [N,64]
  float* bufB  = bufA + N64;         // scatter output / h          [N,64]
  float* dis   = bufB + N64;         // deg -> 1/sqrt(deg)          [N]
  float* normv = dis + N;            // per-edge norm               [E]
  float* stats = normv + E;          // sums[64] | sqs[64]
  float* gx    = stats + 128;        // segment max                 [B,64]

  const int T = 256;
  const long long totN64 = N64;
  const long long totE64 = (long long)E * HCOLS;
  // grid-strided GEMM: several row-tiles per wave amortize the LDS staging of W
  const int gemmGrid = 256;

  // --- degree / norm ---
  k_fill<<<cdiv(N, T), T, 0, stream>>>(dis, 1.0f, N);
  k_deg_edges<<<cdiv(E, T), T, 0, stream>>>(col, ew, dis, E);
  k_rsqrt_inplace<<<cdiv(N, T), T, 0, stream>>>(dis, N);
  k_edge_norm<<<cdiv(E, T), T, 0, stream>>>(row, col, ew, dis, normv, E);

  // --- GCN layer 1 (K = F = 128) ---
  k_gemm_wmma<<<gemmGrid, T, 0, stream>>>(x, W1, bufA, N, F);
  k_init_out<<<cdiv(totN64, T), T, 0, stream>>>(bufA, dis, b1, bufB, totN64);
  k_scatter<<<cdiv(totE64, T), T, 0, stream>>>(bufA, row, col, normv, bufB, totE64);
  k_fill<<<1, 128, 0, stream>>>(stats, 0.f, 128);
  k_colstats<<<256, 256, 0, stream>>>(bufB, N, stats, stats + 64);
  k_bnrelu<<<cdiv(totN64, T), T, 0, stream>>>(bufB, stats, stats + 64, g1, be1, bufB, N, totN64);

  // --- GCN layer 2 (K = H = 64) ---
  k_gemm_wmma<<<gemmGrid, T, 0, stream>>>(bufB, W2, bufA, N, HCOLS);
  k_init_out<<<cdiv(totN64, T), T, 0, stream>>>(bufA, dis, b2, bufB, totN64);
  k_scatter<<<cdiv(totE64, T), T, 0, stream>>>(bufA, row, col, normv, bufB, totE64);
  k_fill<<<1, 128, 0, stream>>>(stats, 0.f, 128);
  k_colstats<<<256, 256, 0, stream>>>(bufB, N, stats, stats + 64);
  k_bnrelu<<<cdiv(totN64, T), T, 0, stream>>>(bufB, stats, stats + 64, g2, be2, bufB, N, totN64);

  // --- global max pool ---
  k_fill<<<cdiv((long long)B * HCOLS, T), T, 0, stream>>>(gx, 0.f, (long long)B * HCOLS);
  k_segmax<<<cdiv(totN64, T), T, 0, stream>>>(bufB, bidx, (unsigned int*)gx, totN64);

  // --- encoders + head ---
  k_head<<<1, 64, 0, stream>>>(gx, speed, route, sw, sb, sg, sbe, cw, cb, rg, rbe,
                               rw, rb, ow1, ob1, og, obe, ow2, ob2, (float*)d_out);
}
